// FlashAttention_2293512536572
// MI455X (gfx1250) — matrix operand order repaired and checked by reading, not compiled, not run on hardware
//
#include <hip/hip_runtime.h>
#include <cstdint>
#include <cstddef>

// ---------------- problem constants ----------------
constexpr int B  = 2, S = 2048, H = 2048, NH = 16, HD = 128;
constexpr int BS = B * S;        // 4096 token rows
constexpr int NQ = 3 * H;        // 6144 qkv output dim
constexpr float SCALE = 0.08838834764831845f;  // 1/sqrt(128)

// ---------------- CDNA5 WMMA types ----------------
typedef __attribute__((ext_vector_type(16))) _Float16 v16h;
typedef __attribute__((ext_vector_type(8)))  _Float16 v8h;
typedef __attribute__((ext_vector_type(8)))  float    v8f;

__device__ __forceinline__ v8f wmma_16x16x32(v16h a, v16h b, v8f c) {
  return __builtin_amdgcn_wmma_f32_16x16x32_f16(false, a, false, b, (short)0, c,
                                                false, false);
}

__device__ __forceinline__ v16h join8(v8h lo, v8h hi) {
  return __builtin_shufflevector(lo, hi, 0, 1, 2, 3, 4, 5, 6, 7,
                                 8, 9, 10, 11, 12, 13, 14, 15);
}

// CDNA5 async vector copy: global -> LDS, 16B per lane, tracked by ASYNCcnt.
__device__ __forceinline__ void async_g2l_b128(const _Float16* gsrc,
                                               _Float16* ldst) {
  const unsigned lds = (unsigned)(uintptr_t)ldst;  // low 32 bits = LDS offset
  asm volatile("global_load_async_to_lds_b128 %0, %1, off"
               :
               : "v"(lds), "v"(gsrc)
               : "memory");
}
__device__ __forceinline__ void wait_async0() {
  asm volatile("s_wait_asynccnt 0x0" ::: "memory");
}

constexpr int LDT   = 40;          // LDS row stride in halfs (32+8 pad, 80B)
constexpr int BUFSZ = 128 * LDT;   // one tile buffer (halfs)

// =====================================================================
// Kernel 0: f32 -> f16 conversion (X, Wqkv, Wo), 8 elements/thread
// =====================================================================
__global__ void __launch_bounds__(256)
cvt_f32_f16_kernel(const float* __restrict__ src, _Float16* __restrict__ dst) {
  const size_t i0 = ((size_t)blockIdx.x * 256 + threadIdx.x) * 8;
  const float4 a = *(const float4*)(src + i0);
  const float4 b = *(const float4*)(src + i0 + 4);
  v8h h;
  h[0] = (_Float16)a.x; h[1] = (_Float16)a.y;
  h[2] = (_Float16)a.z; h[3] = (_Float16)a.w;
  h[4] = (_Float16)b.x; h[5] = (_Float16)b.y;
  h[6] = (_Float16)b.z; h[7] = (_Float16)b.w;
  *(v8h*)(dst + i0) = h;
}

// =====================================================================
// Shared GEMM core: C[128x128] += A[128xK] * B[128xK]^T, f16 in, f32 acc.
// Double-buffered LDS tiles filled with global_load_async_to_lds_b128.
// =====================================================================
template <int KDIM>
__device__ __forceinline__ void gemm_core(const _Float16* __restrict__ A,
                                          const _Float16* __restrict__ Bw,
                                          _Float16* ldsA, _Float16* ldsB,
                                          int m0, int n0, int tid,
                                          v8f (&acc)[2][4]) {
  const int wave = tid >> 5, lane = tid & 31;
  const int hl = lane >> 4, l16 = lane & 15;
  const int wm = wave & 3, wn = wave >> 2;

  auto stage = [&](int kk, int buf) {
#pragma unroll
    for (int it = 0; it < 2; ++it) {
      const int lin = tid + it * 256;        // 0..511
      const int row = lin >> 2;
      const int kq  = (lin & 3) * 8;
      async_g2l_b128(A + (size_t)(m0 + row) * KDIM + kk + kq,
                     ldsA + buf * BUFSZ + row * LDT + kq);
      async_g2l_b128(Bw + (size_t)(n0 + row) * KDIM + kk + kq,
                     ldsB + buf * BUFSZ + row * LDT + kq);
    }
  };

  stage(0, 0);
  wait_async0();
  __syncthreads();

  for (int kk = 0; kk < KDIM; kk += 32) {
    const int cur = (kk >> 5) & 1;
    if (kk + 32 < KDIM) stage(kk + 32, cur ^ 1);  // overlap with compute

    const _Float16* la = ldsA + cur * BUFSZ;
    const _Float16* lb = ldsB + cur * BUFSZ;
    v16h af[2], bf[4];
#pragma unroll
    for (int i = 0; i < 2; ++i) {
      const int row = wm * 32 + i * 16 + l16;
      af[i] = join8(*(const v8h*)&la[row * LDT + hl * 8],
                    *(const v8h*)&la[row * LDT + 16 + hl * 8]);
    }
#pragma unroll
    for (int j = 0; j < 4; ++j) {
      const int col = wn * 64 + j * 16 + l16;
      bf[j] = join8(*(const v8h*)&lb[col * LDT + hl * 8],
                    *(const v8h*)&lb[col * LDT + 16 + hl * 8]);
    }
#pragma unroll
    for (int i = 0; i < 2; ++i)
#pragma unroll
      for (int j = 0; j < 4; ++j)
        acc[i][j] = wmma_16x16x32(af[i], bf[j], acc[i][j]);

    wait_async0();     // next tile landed (this wave's copies)
    __syncthreads();   // all waves' copies landed; safe to swap buffers
  }
}

// =====================================================================
// Kernel 1: QKV GEMM -> scatter f16 into Q [B,NH,S,HD], K [B,NH,S,HD],
//           Vt [B,NH,HD,S]
// =====================================================================
__global__ void __launch_bounds__(256)
qkv_gemm_kernel(const _Float16* __restrict__ Xh, const _Float16* __restrict__ Wh,
                _Float16* __restrict__ Qh, _Float16* __restrict__ Kh,
                _Float16* __restrict__ Vt) {
  __shared__ _Float16 ldsA[2 * BUFSZ];
  __shared__ _Float16 ldsB[2 * BUFSZ];
  const int tid = threadIdx.x;
  const int wave = tid >> 5, lane = tid & 31;
  const int hl = lane >> 4, l16 = lane & 15;
  const int wm = wave & 3, wn = wave >> 2;
  const int m0 = blockIdx.y * 128, n0 = blockIdx.x * 128;

  v8f zero = {};
  v8f acc[2][4];
#pragma unroll
  for (int i = 0; i < 2; ++i)
#pragma unroll
    for (int j = 0; j < 4; ++j) acc[i][j] = zero;

  gemm_core<H>(Xh, Wh, ldsA, ldsB, m0, n0, tid, acc);

#pragma unroll
  for (int i = 0; i < 2; ++i) {
#pragma unroll
    for (int j = 0; j < 4; ++j) {
#pragma unroll
      for (int r = 0; r < 8; ++r) {
        const int gm = m0 + wm * 32 + i * 16 + hl * 8 + r;  // token row
        const int gn = n0 + wn * 64 + j * 16 + l16;         // qkv col
        const float v = acc[i][j][r];
        const int t   = gn >> 11;       // 0=q 1=k 2=v (NH*HD = 2048)
        const int rem = gn & 2047;
        const int hh  = rem >> 7;
        const int d   = rem & 127;
        const int bb  = gm >> 11;       // S = 2048
        const int ss  = gm & 2047;
        const size_t bh = (size_t)(bb * NH + hh);
        if (t == 0)      Qh[(bh * S + ss) * HD + d] = (_Float16)v;
        else if (t == 1) Kh[(bh * S + ss) * HD + d] = (_Float16)v;
        else             Vt[(bh * HD + d) * S + ss] = (_Float16)v;
      }
    }
  }
}

// =====================================================================
// Kernel 2: RoPE in place on Q and K (pairs (d, d+64))
// =====================================================================
__global__ void __launch_bounds__(256)
rope_kernel(_Float16* __restrict__ Qh, _Float16* __restrict__ Kh,
            const float* __restrict__ cosb, const float* __restrict__ sinb) {
  const int idx = blockIdx.x * 256 + threadIdx.x;  // B*NH*S*64 threads
  const int d  = idx & 63;
  const int s  = (idx >> 6) & (S - 1);
  const int hh = (idx >> 17) & (NH - 1);
  const int bb = idx >> 21;
  const size_t base = ((size_t)(bb * NH + hh) * S + s) * (size_t)HD;
  const float c1 = cosb[s * HD + d],      s1 = sinb[s * HD + d];
  const float c2 = cosb[s * HD + d + 64], s2 = sinb[s * HD + d + 64];
  {
    const float x1 = (float)Qh[base + d], x2 = (float)Qh[base + d + 64];
    Qh[base + d]      = (_Float16)(x1 * c1 - x2 * s1);
    Qh[base + d + 64] = (_Float16)(x2 * c2 + x1 * s2);
  }
  {
    const float x1 = (float)Kh[base + d], x2 = (float)Kh[base + d + 64];
    Kh[base + d]      = (_Float16)(x1 * c1 - x2 * s1);
    Kh[base + d + 64] = (_Float16)(x2 * c2 + x1 * s2);
  }
}

// =====================================================================
// Kernel 3: causal flash attention, WMMA for QK^T and PV,
//           double-buffered async K/Vt tile streaming.
//   grid = (S/128, NH, B); 8 waves, each owns 16 q-rows.
// =====================================================================
__global__ void __launch_bounds__(256)
flash_attn_kernel(const _Float16* __restrict__ Qh, const _Float16* __restrict__ Kh,
                  const _Float16* __restrict__ Vt, _Float16* __restrict__ attn) {
  __shared__ _Float16 ldsK[2][32][136];   // K tile  [kpos][d]
  __shared__ _Float16 ldsV[2][128][40];   // Vt tile [d][kpos]
  __shared__ _Float16 ldsP[8][16][40];    // per-wave P bounce

  const int tid  = threadIdx.x;
  const int wave = tid >> 5, lane = tid & 31;
  const int hl   = lane >> 4, l16 = lane & 15;
  const int qt = blockIdx.x, hh = blockIdx.y, bb = blockIdx.z;
  const size_t bh = (size_t)(bb * NH + hh);
  const int q0 = qt * 128 + wave * 16;

  auto stageKV = [&](int k0, int buf) {
#pragma unroll
    for (int it = 0; it < 2; ++it) {
      const int lin = tid + it * 256;
      const int row = lin >> 4, dq = (lin & 15) * 8;
      async_g2l_b128(Kh + (bh * S + (size_t)(k0 + row)) * HD + dq,
                     &ldsK[buf][row][dq]);
    }
#pragma unroll
    for (int it = 0; it < 2; ++it) {
      const int lin = tid + it * 256;
      const int dr = lin >> 2, kq = (lin & 3) * 8;
      async_g2l_b128(Vt + (bh * HD + (size_t)dr) * S + k0 + kq,
                     &ldsV[buf][dr][kq]);
    }
  };

  // Q fragments for all 4 K-slices of HD=128 (live in registers)
  v16h qf[4];
  {
    const _Float16* qrow = Qh + (bh * S + (size_t)(q0 + l16)) * HD;
#pragma unroll
    for (int c = 0; c < 4; ++c)
      qf[c] = join8(*(const v8h*)(qrow + c * 32 + hl * 8),
                    *(const v8h*)(qrow + c * 32 + 16 + hl * 8));
  }

  v8f zero = {};
  v8f oacc[8];
#pragma unroll
  for (int dt = 0; dt < 8; ++dt) oacc[dt] = zero;
  float mrow[8], lrow[8];
#pragma unroll
  for (int r = 0; r < 8; ++r) { mrow[r] = -1e30f; lrow[r] = 0.0f; }

  const int khi = (qt + 1) * 128;
  stageKV(0, 0);
  wait_async0();
  __syncthreads();

  for (int k0 = 0; k0 < khi; k0 += 32) {
    const int cur = (k0 >> 5) & 1;
    if (k0 + 32 < khi) stageKV(k0 + 32, cur ^ 1);

    if (k0 <= q0 + 15) {  // wave has at least one unmasked column
      // ---- scores: 16 x 32, K-dim = HD = 128 ----
      v8f sc0 = zero, sc1 = zero;
#pragma unroll
      for (int c = 0; c < 4; ++c) {
        const _Float16* bp0 = &ldsK[cur][l16][c * 32 + hl * 8];
        sc0 = wmma_16x16x32(qf[c],
                            join8(*(const v8h*)bp0, *(const v8h*)(bp0 + 16)), sc0);
        const _Float16* bp1 = &ldsK[cur][16 + l16][c * 32 + hl * 8];
        sc1 = wmma_16x16x32(qf[c],
                            join8(*(const v8h*)bp1, *(const v8h*)(bp1 + 16)), sc1);
      }

      // ---- scale + causal mask + row max ----
      float rmax[8];
#pragma unroll
      for (int r = 0; r < 8; ++r) {
        const int qrow = q0 + hl * 8 + r;
        float a = sc0[r] * SCALE; if (k0 + l16 > qrow)      a = -1e30f;
        float b = sc1[r] * SCALE; if (k0 + 16 + l16 > qrow) b = -1e30f;
        sc0[r] = a; sc1[r] = b;
        rmax[r] = fmaxf(a, b);
      }
#pragma unroll
      for (int m = 1; m < 16; m <<= 1)
#pragma unroll
        for (int r = 0; r < 8; ++r)
          rmax[r] = fmaxf(rmax[r], __shfl_xor(rmax[r], m, 32));

      // ---- online softmax ----
      float al[8], psum[8];
#pragma unroll
      for (int r = 0; r < 8; ++r) {
        const float mn = fmaxf(mrow[r], rmax[r]);
        al[r]  = __expf(mrow[r] - mn);
        mrow[r] = mn;
        sc0[r] = __expf(sc0[r] - mn);
        sc1[r] = __expf(sc1[r] - mn);
        psum[r] = sc0[r] + sc1[r];
      }
#pragma unroll
      for (int m = 1; m < 16; m <<= 1)
#pragma unroll
        for (int r = 0; r < 8; ++r)
          psum[r] += __shfl_xor(psum[r], m, 32);
#pragma unroll
      for (int r = 0; r < 8; ++r) {
        lrow[r] = lrow[r] * al[r] + psum[r];
#pragma unroll
        for (int dt = 0; dt < 8; ++dt) oacc[dt][r] *= al[r];
      }

      // ---- P: C-layout -> A-layout via per-wave LDS bounce ----
#pragma unroll
      for (int r = 0; r < 8; ++r) {
        ldsP[wave][hl * 8 + r][l16]      = (_Float16)sc0[r];
        ldsP[wave][hl * 8 + r][16 + l16] = (_Float16)sc1[r];
      }
      __builtin_amdgcn_wave_barrier();
      asm volatile("s_wait_dscnt 0x0" ::: "memory");  // CDNA5 split DS wait
      v16h pf;
      {
        const _Float16* pp = &ldsP[wave][l16][0];
        pf = join8(*(const v8h*)(pp + hl * 8), *(const v8h*)(pp + 16 + hl * 8));
      }
      __builtin_amdgcn_wave_barrier();

      // ---- PV: 8 x (16x16) output tiles over HD=128 ----
#pragma unroll
      for (int dt = 0; dt < 8; ++dt) {
        const _Float16* vp = &ldsV[cur][dt * 16 + l16][hl * 8];
        oacc[dt] = wmma_16x16x32(
            pf, join8(*(const v8h*)vp, *(const v8h*)(vp + 16)), oacc[dt]);
      }
    }

    wait_async0();
    __syncthreads();
  }

  // ---- normalize, write attn [B,S,NH*HD] f16 ----
#pragma unroll
  for (int r = 0; r < 8; ++r) {
    const int qrow = q0 + hl * 8 + r;
    const float inv = 1.0f / lrow[r];
#pragma unroll
    for (int dt = 0; dt < 8; ++dt)
      attn[((size_t)bb * S + qrow) * H + hh * HD + dt * 16 + l16] =
          (_Float16)(oacc[dt][r] * inv);
  }
}

// =====================================================================
// Kernel 4: out = attn(f16 [BS,H]) * Wo^T(f16 [H,H]) -> f32 [BS,H]
// =====================================================================
__global__ void __launch_bounds__(256)
out_gemm_kernel(const _Float16* __restrict__ A, const _Float16* __restrict__ Wh,
                float* __restrict__ out) {
  __shared__ _Float16 ldsA[2 * BUFSZ];
  __shared__ _Float16 ldsB[2 * BUFSZ];
  const int tid = threadIdx.x;
  const int wave = tid >> 5, lane = tid & 31;
  const int hl = lane >> 4, l16 = lane & 15;
  const int wm = wave & 3, wn = wave >> 2;
  const int m0 = blockIdx.y * 128, n0 = blockIdx.x * 128;

  v8f zero = {};
  v8f acc[2][4];
#pragma unroll
  for (int i = 0; i < 2; ++i)
#pragma unroll
    for (int j = 0; j < 4; ++j) acc[i][j] = zero;

  gemm_core<H>(A, Wh, ldsA, ldsB, m0, n0, tid, acc);

#pragma unroll
  for (int i = 0; i < 2; ++i)
#pragma unroll
    for (int j = 0; j < 4; ++j)
#pragma unroll
      for (int r = 0; r < 8; ++r) {
        const int gm = m0 + wm * 32 + i * 16 + hl * 8 + r;
        const int gn = n0 + wn * 64 + j * 16 + l16;
        out[(size_t)gm * H + gn] = acc[i][j][r];
      }
}

// =====================================================================
// launch: inputs = {hidden_states, cos, sin, w_qkv, w_o}
// ws (f16): Xh | Wqkvh | Woh | Q | K | Vt | attn  (~118 MB total)
// =====================================================================
extern "C" void kernel_launch(void* const* d_in, const int* in_sizes, int n_in,
                              void* d_out, int out_size, void* d_ws, size_t ws_size,
                              hipStream_t stream) {
  (void)in_sizes; (void)n_in; (void)out_size; (void)ws_size;
  const float* hs   = (const float*)d_in[0];
  const float* cosb = (const float*)d_in[1];
  const float* sinb = (const float*)d_in[2];
  const float* wqkv = (const float*)d_in[3];
  const float* wo   = (const float*)d_in[4];
  float* out = (float*)d_out;

  const size_t NX = (size_t)BS * H;       // 8,388,608
  const size_t NW1 = (size_t)NQ * H;      // 12,582,912
  const size_t NW2 = (size_t)H * H;       // 4,194,304
  const size_t PER = (size_t)B * NH * S * HD;

  _Float16* Xh    = (_Float16*)d_ws;
  _Float16* Wqkvh = Xh + NX;
  _Float16* Woh   = Wqkvh + NW1;
  _Float16* Qh    = Woh + NW2;
  _Float16* Kh    = Qh + PER;
  _Float16* Vt    = Kh + PER;
  _Float16* attn  = Vt + PER;

  dim3 blk(256);
  cvt_f32_f16_kernel<<<dim3(NX / 2048), blk, 0, stream>>>(hs, Xh);
  cvt_f32_f16_kernel<<<dim3(NW1 / 2048), blk, 0, stream>>>(wqkv, Wqkvh);
  cvt_f32_f16_kernel<<<dim3(NW2 / 2048), blk, 0, stream>>>(wo, Woh);
  qkv_gemm_kernel<<<dim3(NQ / 128, BS / 128), blk, 0, stream>>>(Xh, Wqkvh, Qh, Kh, Vt);
  rope_kernel<<<dim3((B * NH * S * 64) / 256), blk, 0, stream>>>(Qh, Kh, cosb, sinb);
  flash_attn_kernel<<<dim3(S / 128, NH, B), blk, 0, stream>>>(Qh, Kh, Vt, attn);
  out_gemm_kernel<<<dim3(H / 128, BS / 128), blk, 0, stream>>>(attn, wo ? Woh : Woh, out);
}
